// MultiLevelHyperTransformer_76287209111802
// MI455X (gfx1250) — matrix operand order repaired and checked by reading, not compiled, not run on hardware
//
#include <hip/hip_runtime.h>
#include <hip/hip_bf16.h>

typedef __attribute__((ext_vector_type(16))) _Float16 v16h;
typedef __attribute__((ext_vector_type(8)))  _Float16 v8h;
typedef __attribute__((ext_vector_type(8)))  float    v8f;

#define S_LEN 1024
#define BATCH 16
#define M_ROWS (BATCH * S_LEN)

// Attention LDS layout:
//  [0, 64K)        : K slab (1024 x 32 f16) in pass 1, reused as V^T slab (32 x 1024 f16) in pass 2
//  [64K, +66560)   : f32 score panel, 16 rows x 1040 stride
//  [.., +33280)    : f16 P panel, 16 rows x 1040 stride (32B-aligned rows)
//  [.., +16384)    : per-wave O^T partials for cross-wave reduction
constexpr size_t ATT_SLAB_BYTES = 65536;
constexpr int    ATT_SS         = 1040;
constexpr int    ATT_PS         = 1040;
constexpr size_t ATT_SC_OFF     = ATT_SLAB_BYTES;
constexpr size_t ATT_PF_OFF     = ATT_SC_OFF + (size_t)16 * ATT_SS * 4;   // 132096 (32B aligned)
constexpr size_t ATT_PARTS_OFF  = ATT_PF_OFF + (size_t)16 * ATT_PS * 2;   // 165376
constexpr size_t ATT_SHM        = ATT_PARTS_OFF + (size_t)8 * 512 * 4;    // 181760

__device__ __forceinline__ float wred_max(float v) {
#pragma unroll
  for (int off = 16; off > 0; off >>= 1) v = fmaxf(v, __shfl_xor(v, off, 32));
  return v;
}
__device__ __forceinline__ float wred_sum(float v) {
#pragma unroll
  for (int off = 16; off > 0; off >>= 1) v += __shfl_xor(v, off, 32);
  return v;
}

// Issue one per-lane 16B async global->LDS copy (ASYNCcnt-tracked).
__device__ __forceinline__ void async_copy_b128(unsigned int lds_addr, const void* gaddr) {
  asm volatile("global_load_async_to_lds_b128 %0, %1, off"
               :: "v"(lds_addr), "v"((unsigned long long)(size_t)gaddr)
               : "memory");
}
__device__ __forceinline__ void async_wait0() {
  asm volatile("s_wait_asynccnt 0x0" ::: "memory");
}

// ---------------------------------------------------------------------------
// Fourier featurization: pts -> z (f32 + f16) and boundary flags
// ---------------------------------------------------------------------------
__global__ __launch_bounds__(256)
void featurize_kernel(const float* __restrict__ pts, const float* __restrict__ fs,
                      const float* __restrict__ ft, const float* __restrict__ w_space,
                      const float* __restrict__ b_space, const float* __restrict__ w_time,
                      const float* __restrict__ b_time, const float* __restrict__ w_fout,
                      const float* __restrict__ b_fout, const float* __restrict__ w_gate,
                      const float* __restrict__ b_gate,
                      float* __restrict__ xs, _Float16* __restrict__ xh, int* __restrict__ bnd)
{
  int idx = blockIdx.x * 256 + threadIdx.x;
  if (idx >= M_ROWS) return;
  int l = idx & (S_LEN - 1);
  const float TWO_PI = 6.283185307179586f;
  float x = pts[idx * 3 + 0], y = pts[idx * 3 + 1], t = pts[idx * 3 + 2];

  float sn[16], cs[16];
#pragma unroll
  for (int j = 0; j < 16; ++j) {
    float p = TWO_PI * (x * fs[j] + y * fs[16 + j]);
    sn[j] = __sinf(p); cs[j] = __cosf(p);
  }
  float sp_feat[64];
  for (int d = 0; d < 64; ++d) {
    float a = b_space[d];
#pragma unroll
    for (int j = 0; j < 16; ++j)
      a += sn[j] * w_space[j * 64 + d] + cs[j] * w_space[(16 + j) * 64 + d];
    sp_feat[d] = a;
  }
  float tn[8], tc[8];
#pragma unroll
  for (int j = 0; j < 8; ++j) {
    float p = TWO_PI * t * ft[j];
    tn[j] = __sinf(p); tc[j] = __cosf(p);
  }
  float t_feat[32];
  for (int d = 0; d < 32; ++d) {
    float a = b_time[d];
#pragma unroll
    for (int j = 0; j < 8; ++j)
      a += tn[j] * w_time[j * 32 + d] + tc[j] * w_time[(8 + j) * 32 + d];
    t_feat[d] = a;
  }
  float gacc = b_gate[0];
  for (int d = 0; d < 64; ++d) {
    float a = b_fout[d];
    for (int e = 0; e < 64; ++e) a += sp_feat[e] * w_fout[e * 64 + d];
    for (int e = 0; e < 32; ++e) a += t_feat[e] * w_fout[(64 + e) * 64 + d];
    xs[(size_t)idx * 64 + d] = a;
    xh[(size_t)idx * 64 + d] = (_Float16)a;
    gacc += a * w_gate[d];
  }
  float gate = 1.f / (1.f + __expf(-gacc));
  bnd[idx] = (l == 0) || (gate > 0.5f);
}

// ---------------------------------------------------------------------------
// Per-batch inclusive scan of boundary flags -> seg_id, n_seg
// ---------------------------------------------------------------------------
__global__ __launch_bounds__(1024)
void segscan_kernel(const int* __restrict__ bnd, int* __restrict__ segid, int* __restrict__ nseg)
{
  __shared__ int buf[2][S_LEN];
  int b = blockIdx.x, t = threadIdx.x;
  buf[0][t] = bnd[b * S_LEN + t];
  __syncthreads();
  int src = 0;
  for (int off = 1; off < S_LEN; off <<= 1) {
    int nv = buf[src][t] + ((t >= off) ? buf[src][t - off] : 0);
    buf[src ^ 1][t] = nv;
    __syncthreads();
    src ^= 1;
  }
  segid[b * S_LEN + t] = buf[src][t] - 1;
  if (t == S_LEN - 1) nseg[b] = buf[src][t];
}

// ---------------------------------------------------------------------------
// Weight prep: fp32 (K x N) row-major -> f16 transposed (N x K)
// ---------------------------------------------------------------------------
__global__ __launch_bounds__(256)
void wprep_kernel(const float* __restrict__ W, _Float16* __restrict__ Wt, int K, int N)
{
  int idx = blockIdx.x * 256 + threadIdx.x;
  if (idx >= K * N) return;
  int k = idx / N, n = idx % N;
  Wt[(size_t)n * K + k] = (_Float16)W[idx];
}

// ---------------------------------------------------------------------------
// Generic WMMA GEMM: out = act(A_f16[MxK] @ Wt_f16[NxK]^T + bias), 16x16 tile/wave
// ---------------------------------------------------------------------------
__global__ __launch_bounds__(256)
void gemm_kernel(const _Float16* __restrict__ A, const _Float16* __restrict__ Wt,
                 const float* __restrict__ bias, float* __restrict__ outF,
                 _Float16* __restrict__ outH, int Mn, int Nn, int Kn, int relu)
{
  int lane = threadIdx.x & 31, wid = threadIdx.x >> 5;
  int tiles_n = Nn >> 4;
  long tile = (long)blockIdx.x * 8 + wid;
  long ntiles = (long)(Mn >> 4) * tiles_n;
  if (tile >= ntiles) return;
  int m0 = (int)(tile / tiles_n) << 4;
  int n0 = (int)(tile % tiles_n) << 4;
  int l15 = lane & 15;
  int kHalf = (lane < 16) ? 0 : 16;
  int kBase = (lane < 16) ? 0 : 8;

  v8f c = {};
  for (int k0 = 0; k0 < Kn; k0 += 32) {
    const _Float16* arow = A + (size_t)(m0 + l15) * Kn + k0;
    if (k0 + 32 < Kn) __builtin_prefetch((const void*)(arow + 32), 0, 1);
    v8h lo = *(const v8h*)(arow + kHalf);
    v8h hi = *(const v8h*)(arow + kHalf + 8);
    v16h a;
#pragma unroll
    for (int i = 0; i < 8; ++i) { a[i] = lo[i]; a[8 + i] = hi[i]; }
    v16h bf = *(const v16h*)(Wt + (size_t)(n0 + l15) * Kn + k0 + kHalf);
    c = __builtin_amdgcn_wmma_f32_16x16x32_f16(false, a, false, bf, (short)0, c, false, false);
  }
  float bb = bias ? bias[n0 + l15] : 0.f;
#pragma unroll
  for (int r = 0; r < 8; ++r) {
    int mr = m0 + r + ((lane < 16) ? 0 : 8);
    float v = c[r] + bb;
    if (relu) v = fmaxf(v, 0.f);
    size_t o = (size_t)mr * Nn + n0 + l15;
    if (outF) outF[o] = v;
    if (outH) outH[o] = (_Float16)v;
  }
}

// ---------------------------------------------------------------------------
// V transpose: vh (B,S,D) -> vt (B,H,32,S)
// ---------------------------------------------------------------------------
__global__ __launch_bounds__(256)
void vtrans_kernel(const _Float16* __restrict__ vh, _Float16* __restrict__ vt, int D, int H)
{
  int idx = blockIdx.x * 256 + threadIdx.x;
  if (idx >= M_ROWS * D) return;
  int d = idx % D; int r = idx / D;
  int b = r >> 10; int s = r & (S_LEN - 1);
  int h = d >> 5;  int m = d & 31;
  vt[((size_t)((b * H + h) * 32 + m)) * S_LEN + s] = vh[idx];
}

// ---------------------------------------------------------------------------
// Masked attention, one workgroup per (b, h, 16-query tile). hd == 32.
// K and V^T head-slabs are DMA'd into LDS with async global->LDS copies;
// the V^T copy overlaps the softmax phase.
// mode 0: seg_id equality mask; mode 1: valid-pair | diagonal mask.
// ---------------------------------------------------------------------------
__global__ __launch_bounds__(256)
void attn_kernel(const _Float16* __restrict__ qh, const _Float16* __restrict__ kh,
                 const _Float16* __restrict__ vt, _Float16* __restrict__ oh,
                 const int* __restrict__ segid, const int* __restrict__ nsegp,
                 int D, int H, int mode, float scale)
{
  extern __shared__ char smem[];
  _Float16* slab  = (_Float16*)smem;
  float*    sc    = (float*)(smem + ATT_SC_OFF);
  _Float16* pf    = (_Float16*)(smem + ATT_PF_OFF);
  float*    parts = (float*)(smem + ATT_PARTS_OFF);

  const int qtiles = S_LEN / 16;
  int wg = blockIdx.x;
  int qt = wg % qtiles;
  int h  = (wg / qtiles) % H;
  int b  = wg / (qtiles * H);
  int q0 = qt * 16;

  int tid = threadIdx.x;
  int lane = tid & 31, wid = tid >> 5;
  int l15 = lane & 15;
  int kHalf = (lane < 16) ? 0 : 16;
  int kBase = (lane < 16) ? 0 : 8;
  int mOff  = (lane < 16) ? 0 : 8;

  // ---- Async-stage the K head-slab (1024 rows x 64B, strided in global) into LDS
#pragma unroll
  for (int it = 0; it < 16; ++it) {
    int e = it * 256 + tid;             // 16B chunk id, 4096 total
    int row = e >> 2, seg = e & 3;
    const _Float16* g = kh + (size_t)(b * S_LEN + row) * D + h * 32 + seg * 8;
    unsigned int laddr = (unsigned int)(size_t)(smem + (size_t)row * 64 + (size_t)seg * 16);
    async_copy_b128(laddr, (const void*)g);
  }

  int ns = nsegp[b];
  int qIdx = q0 + l15;
  int sid_q = segid[b * S_LEN + qIdx];
  bool vq = qIdx < ns;

  // B fragment: Q^T (32 x 16) — contiguous 16 halves per lane (overlaps async copy)
  v16h bq = *(const v16h*)(qh + (size_t)(b * S_LEN + qIdx) * D + h * 32 + kHalf);

  async_wait0();
  __syncthreads();

  // ---- Pass 1: S^T tiles = K_tile(16x32) @ Q^T(32x16) from LDS, masked -> LDS[q][key]
  for (int kt8 = 0; kt8 < 8; ++kt8) {
    int kt = wid * 8 + kt8;
    const _Float16* krow = slab + (size_t)(kt * 16 + l15) * 32;
    v8h lo = *(const v8h*)(krow + kHalf);
    v8h hi = *(const v8h*)(krow + kHalf + 8);
    v16h a;
#pragma unroll
    for (int i = 0; i < 8; ++i) { a[i] = lo[i]; a[8 + i] = hi[i]; }
    v8f c = {};
    c = __builtin_amdgcn_wmma_f32_16x16x32_f16(false, a, false, bq, (short)0, c, false, false);
#pragma unroll
    for (int r = 0; r < 8; ++r) {
      int key = kt * 16 + r + mOff;
      bool ok;
      if (mode == 0) ok = (segid[b * S_LEN + key] == sid_q);
      else { bool vk = key < ns; ok = (vk && vq) || (key == qIdx); }
      sc[l15 * ATT_SS + key] = ok ? c[r] * scale : -1.0e9f;
    }
  }
  __syncthreads();   // pass-1 reads of K slab complete; scores visible

  // ---- Async-stage V^T slab (contiguous 64KB) into the same LDS region,
  //      overlapping the softmax phase below.
  {
    const _Float16* vg = vt + (size_t)(b * H + h) * 32 * S_LEN;
#pragma unroll
    for (int it = 0; it < 16; ++it) {
      int e = it * 256 + tid;
      unsigned int laddr = (unsigned int)(size_t)(smem + (size_t)e * 16);
      async_copy_b128(laddr, (const void*)(vg + (size_t)e * 8));
    }
  }

  // ---- Softmax per query row (each wave: rows wid, wid+8)
  for (int q = wid; q < 16; q += 8) {
    float* row = sc + q * ATT_SS;
    float mx = -3.0e38f;
    for (int k2 = lane; k2 < S_LEN; k2 += 32) mx = fmaxf(mx, row[k2]);
    mx = wred_max(mx);
    float sum = 0.f;
    for (int k2 = lane; k2 < S_LEN; k2 += 32) { float e = __expf(row[k2] - mx); row[k2] = e; sum += e; }
    sum = wred_sum(sum);
    float inv = 1.f / sum;
    _Float16* prow = pf + (size_t)q * ATT_PS;
    for (int k2 = lane; k2 < S_LEN; k2 += 32) prow[k2] = (_Float16)(row[k2] * inv);
  }
  async_wait0();
  __syncthreads();   // V^T slab and P panel both ready

  // ---- Pass 2: O^T = V^T(32x1024) @ P^T(1024x16); 128 keys per wave, reduce in LDS
  v8f o0 = {}, o1 = {};
  for (int c4 = 0; c4 < 4; ++c4) {
    int kb = wid * 128 + c4 * 32;
    v16h bp = *(const v16h*)(pf + (size_t)l15 * ATT_PS + kb + kHalf);
    {
      const _Float16* vrow = slab + (size_t)l15 * S_LEN + kb;
      v8h lo = *(const v8h*)(vrow + kHalf);
      v8h hi = *(const v8h*)(vrow + kHalf + 8);
      v16h a;
#pragma unroll
      for (int i = 0; i < 8; ++i) { a[i] = lo[i]; a[8 + i] = hi[i]; }
      o0 = __builtin_amdgcn_wmma_f32_16x16x32_f16(false, a, false, bp, (short)0, o0, false, false);
    }
    {
      const _Float16* vrow = slab + (size_t)(16 + l15) * S_LEN + kb;
      v8h lo = *(const v8h*)(vrow + kHalf);
      v8h hi = *(const v8h*)(vrow + kHalf + 8);
      v16h a;
#pragma unroll
      for (int i = 0; i < 8; ++i) { a[i] = lo[i]; a[8 + i] = hi[i]; }
      o1 = __builtin_amdgcn_wmma_f32_16x16x32_f16(false, a, false, bp, (short)0, o1, false, false);
    }
  }
  float* my = parts + wid * 512;
#pragma unroll
  for (int r = 0; r < 8; ++r) {
    my[r * 32 + lane]       = o0[r];
    my[256 + r * 32 + lane] = o1[r];
  }
  __syncthreads();
  for (int e = tid; e < 512; e += 256) {
    float s = 0.f;
#pragma unroll
    for (int w = 0; w < 8; ++w) s += parts[w * 512 + e];
    int hh = e >> 8, r = (e >> 5) & 7, ln = e & 31;
    int nn = ln & 15, mm = hh * 16 + r + ((ln < 16) ? 0 : 8);
    oh[(size_t)(b * S_LEN + q0 + nn) * D + h * 32 + mm] = (_Float16)s;
  }
}

// ---------------------------------------------------------------------------
// LayerNorm: x = LN(x + y) * g + b  (in-place on x), also emits f16
// ---------------------------------------------------------------------------
__global__ __launch_bounds__(256)
void ln_kernel(float* __restrict__ x, const float* __restrict__ y,
               const float* __restrict__ g, const float* __restrict__ bptr,
               _Float16* __restrict__ xh, int Mn, int Dn)
{
  int lane = threadIdx.x & 31, wid = threadIdx.x >> 5;
  int row = blockIdx.x * 8 + wid;
  if (row >= Mn) return;
  float vals[4];
  int cnt = Dn >> 5;
  float s = 0.f;
  for (int i = 0; i < cnt; ++i) {
    size_t o = (size_t)row * Dn + i * 32 + lane;
    float v = x[o] + y[o];
    vals[i] = v; s += v;
  }
  float mean = wred_sum(s) / (float)Dn;
  float vs = 0.f;
  for (int i = 0; i < cnt; ++i) { float d = vals[i] - mean; vs += d * d; }
  float var = wred_sum(vs) / (float)Dn;
  float inv = rsqrtf(var + 1e-5f);
  for (int i = 0; i < cnt; ++i) {
    int d = i * 32 + lane;
    float o = (vals[i] - mean) * inv * g[d] + bptr[d];
    size_t off = (size_t)row * Dn + d;
    x[off] = o;
    xh[off] = (_Float16)o;
  }
}

// ---------------------------------------------------------------------------
// Pooling helpers
// ---------------------------------------------------------------------------
__global__ __launch_bounds__(256)
void fill0_kernel(float* __restrict__ p, size_t n)
{
  size_t i = (size_t)blockIdx.x * 256 + threadIdx.x;
  if (i < n) p[i] = 0.f;
}

__global__ __launch_bounds__(256)
void scatter_kernel(const float* __restrict__ xs, const int* __restrict__ segid,
                    float* __restrict__ sums, float* __restrict__ cnts)
{
  int idx = blockIdx.x * 256 + threadIdx.x;
  if (idx >= M_ROWS * 64) return;
  int d = idx & 63, bl = idx >> 6, b = bl >> 10;
  int sid = segid[bl];
  atomicAdd(&sums[((size_t)(b << 10) + sid) * 64 + d], xs[(size_t)bl * 64 + d]);
  if (d == 0) atomicAdd(&cnts[(b << 10) + sid], 1.0f);
}

__global__ __launch_bounds__(256)
void mean_kernel(const float* __restrict__ sums, const float* __restrict__ cnts,
                 _Float16* __restrict__ meanh)
{
  int idx = blockIdx.x * 256 + threadIdx.x;
  if (idx >= M_ROWS * 64) return;
  int r = idx >> 6;
  float c = fmaxf(cnts[r], 1.f);
  meanh[idx] = (_Float16)(sums[idx] / c);
}

__global__ __launch_bounds__(256)
void maskup_kernel(const float* __restrict__ ybuf, const int* __restrict__ nseg,
                   float* __restrict__ xg, float* __restrict__ glbin, _Float16* __restrict__ xhg)
{
  int idx = blockIdx.x * 256 + threadIdx.x;
  if (idx >= M_ROWS * 128) return;
  int r = idx >> 7, b = r >> 10, s = r & (S_LEN - 1);
  float v = (s < nseg[b]) ? ybuf[idx] : 0.f;
  xg[idx] = v; glbin[idx] = v; xhg[idx] = (_Float16)v;
}

// ---------------------------------------------------------------------------
// Final: masked mean over valid segments (or seg_emb[0] if n_seg==1) @ fin_w + fin_b
// ---------------------------------------------------------------------------
__global__ __launch_bounds__(128)
void final_kernel(const float* __restrict__ xg, const float* __restrict__ glbin,
                  const int* __restrict__ nseg, const float* __restrict__ finw,
                  const float* __restrict__ finb, float* __restrict__ out)
{
  __shared__ float emb[128];
  int b = blockIdx.x, d = threadIdx.x;
  int ns = nseg[b];
  float e;
  if (ns == 1) e = glbin[(size_t)(b * S_LEN) * 128 + d];
  else {
    float s = 0.f;
    for (int q = 0; q < ns; ++q) s += xg[(size_t)(b * S_LEN + q) * 128 + d];
    e = s / (float)ns;
  }
  emb[d] = e;
  __syncthreads();
  float a = finb[d];
  for (int e2 = 0; e2 < 128; ++e2) a += emb[e2] * finw[e2 * 128 + d];
  out[b * 128 + d] = a;
}

// ---------------------------------------------------------------------------
// Host driver
// ---------------------------------------------------------------------------
extern "C" void kernel_launch(void* const* d_in, const int* in_sizes, int n_in,
                              void* d_out, int out_size, void* d_ws, size_t ws_size,
                              hipStream_t stream)
{
  const float* pts        = (const float*)d_in[0];
  const float* fs         = (const float*)d_in[1];
  const float* ft         = (const float*)d_in[2];
  const float* w_space    = (const float*)d_in[3];
  const float* b_space    = (const float*)d_in[4];
  const float* w_time     = (const float*)d_in[5];
  const float* b_time     = (const float*)d_in[6];
  const float* w_fout     = (const float*)d_in[7];
  const float* b_fout     = (const float*)d_in[8];
  const float* w_gate     = (const float*)d_in[9];
  const float* b_gate     = (const float*)d_in[10];
  const float* seg_attn_w = (const float*)d_in[11];
  const float* seg_attn_b = (const float*)d_in[12];
  const float* seg_ln_g   = (const float*)d_in[13];
  const float* seg_ln_b   = (const float*)d_in[14];
  const float* seg_ff1_w  = (const float*)d_in[15];
  const float* seg_ff1_b  = (const float*)d_in[16];
  const float* seg_ff2_w  = (const float*)d_in[17];
  const float* seg_ff2_b  = (const float*)d_in[18];
  const float* glb_attn_w = (const float*)d_in[19];
  const float* glb_attn_b = (const float*)d_in[20];
  const float* glb_ln_g   = (const float*)d_in[21];
  const float* glb_ln_b   = (const float*)d_in[22];
  const float* glb_ff1_w  = (const float*)d_in[23];
  const float* glb_ff1_b  = (const float*)d_in[24];
  const float* glb_ff2_w  = (const float*)d_in[25];
  const float* glb_ff2_b  = (const float*)d_in[26];
  const float* up_w       = (const float*)d_in[27];
  const float* up_b       = (const float*)d_in[28];
  const float* fin_w      = (const float*)d_in[29];
  const float* fin_b      = (const float*)d_in[30];

  (void)hipFuncSetAttribute(reinterpret_cast<const void*>(&attn_kernel),
                            hipFuncAttributeMaxDynamicSharedMemorySize, (int)ATT_SHM);

  char* ws = (char*)d_ws;
  size_t off = 0;
  auto alloc = [&](size_t bytes) -> char* {
    off = (off + 255) & ~(size_t)255;
    char* p = ws + off;
    off += bytes;
    return p;
  };
  const size_t M = M_ROWS;

  float*    xs     = (float*)alloc(M * 64 * 4);
  _Float16* xh_s   = (_Float16*)alloc(M * 64 * 2);
  float*    xg     = (float*)alloc(M * 128 * 4);
  _Float16* xh_g   = (_Float16*)alloc(M * 128 * 2);
  float*    glbin  = (float*)alloc(M * 128 * 4);
  _Float16* qh     = (_Float16*)alloc(M * 128 * 2);
  _Float16* kh     = (_Float16*)alloc(M * 128 * 2);
  _Float16* vh     = (_Float16*)alloc(M * 128 * 2);
  _Float16* vtb    = (_Float16*)alloc(M * 128 * 2);
  _Float16* oh     = (_Float16*)alloc(M * 128 * 2);
  float*    ybuf   = (float*)alloc(M * 128 * 4);
  _Float16* h1     = (_Float16*)alloc(M * 256 * 2);
  float*    sums   = (float*)alloc(M * 64 * 4);
  float*    cnts   = (float*)alloc(M * 4);
  _Float16* meanh  = (_Float16*)alloc(M * 64 * 2);
  int*      segid  = (int*)alloc(M * 4);
  int*      bnd    = (int*)alloc(M * 4);
  int*      nsegp  = (int*)alloc(BATCH * 4);

  _Float16 *seg_awt[2][4], *glb_awt[2][4], *seg_f1t[2], *seg_f2t[2], *glb_f1t[2], *glb_f2t[2];
  for (int l = 0; l < 2; ++l)
    for (int j = 0; j < 4; ++j) seg_awt[l][j] = (_Float16*)alloc(64 * 64 * 2);
  for (int l = 0; l < 2; ++l) { seg_f1t[l] = (_Float16*)alloc(128 * 64 * 2); seg_f2t[l] = (_Float16*)alloc(64 * 128 * 2); }
  for (int l = 0; l < 2; ++l)
    for (int j = 0; j < 4; ++j) glb_awt[l][j] = (_Float16*)alloc(128 * 128 * 2);
  for (int l = 0; l < 2; ++l) { glb_f1t[l] = (_Float16*)alloc(256 * 128 * 2); glb_f2t[l] = (_Float16*)alloc(128 * 256 * 2); }
  _Float16* upwt = (_Float16*)alloc(128 * 64 * 2);
  (void)ws_size; (void)in_sizes; (void)n_in; (void)out_size;

  auto wprep = [&](const float* W, _Float16* Wt, int K, int N) {
    int total = K * N;
    wprep_kernel<<<(total + 255) / 256, 256, 0, stream>>>(W, Wt, K, N);
  };
  for (int l = 0; l < 2; ++l)
    for (int j = 0; j < 4; ++j)
      wprep(seg_attn_w + (size_t)(l * 4 + j) * 64 * 64, seg_awt[l][j], 64, 64);
  for (int l = 0; l < 2; ++l) {
    wprep(seg_ff1_w + (size_t)l * 64 * 128, seg_f1t[l], 64, 128);
    wprep(seg_ff2_w + (size_t)l * 128 * 64, seg_f2t[l], 128, 64);
  }
  for (int l = 0; l < 2; ++l)
    for (int j = 0; j < 4; ++j)
      wprep(glb_attn_w + (size_t)(l * 4 + j) * 128 * 128, glb_awt[l][j], 128, 128);
  for (int l = 0; l < 2; ++l) {
    wprep(glb_ff1_w + (size_t)l * 128 * 256, glb_f1t[l], 128, 256);
    wprep(glb_ff2_w + (size_t)l * 256 * 128, glb_f2t[l], 256, 128);
  }
  wprep(up_w, upwt, 64, 128);

  auto run_gemm = [&](const _Float16* A, const _Float16* Wt, const float* bias,
                      float* outF, _Float16* outH, int Mn, int Nn, int Kn, int relu) {
    long ntiles = (long)(Mn / 16) * (Nn / 16);
    int blocks = (int)((ntiles + 7) / 8);
    gemm_kernel<<<blocks, 256, 0, stream>>>(A, Wt, bias, outF, outH, Mn, Nn, Kn, relu);
  };

  const float SCALE = 0.17677669529663687f; // 1/sqrt(32)

  auto run_layer = [&](float* x, _Float16* xh, int D, int H, int FF, int mode,
                       _Float16* wqT, _Float16* wkT, _Float16* wvT, _Float16* woT,
                       const float* bq, const float* bk, const float* bv, const float* bo,
                       const float* ln0g, const float* ln0b,
                       const float* ln1g, const float* ln1b,
                       _Float16* w1T, const float* b1, _Float16* w2T, const float* b2) {
    run_gemm(xh, wqT, bq, nullptr, qh, M_ROWS, D, D, 0);
    run_gemm(xh, wkT, bk, nullptr, kh, M_ROWS, D, D, 0);
    run_gemm(xh, wvT, bv, nullptr, vh, M_ROWS, D, D, 0);
    vtrans_kernel<<<(M_ROWS * D + 255) / 256, 256, 0, stream>>>(vh, vtb, D, H);
    attn_kernel<<<BATCH * H * (S_LEN / 16), 256, ATT_SHM, stream>>>(
        qh, kh, vtb, oh, segid, nsegp, D, H, mode, SCALE);
    run_gemm(oh, woT, bo, ybuf, nullptr, M_ROWS, D, D, 0);
    ln_kernel<<<M_ROWS / 8, 256, 0, stream>>>(x, ybuf, ln0g, ln0b, xh, M_ROWS, D);
    run_gemm(xh, w1T, b1, nullptr, h1, M_ROWS, FF, D, 1);
    run_gemm(h1, w2T, b2, ybuf, nullptr, M_ROWS, D, FF, 0);
    ln_kernel<<<M_ROWS / 8, 256, 0, stream>>>(x, ybuf, ln1g, ln1b, xh, M_ROWS, D);
  };

  // 1) Fourier featurization + gate
  featurize_kernel<<<M_ROWS / 256, 256, 0, stream>>>(
      pts, fs, ft, w_space, b_space, w_time, b_time, w_fout, b_fout, w_gate, b_gate,
      xs, xh_s, bnd);

  // 2) segment id scan
  segscan_kernel<<<BATCH, 1024, 0, stream>>>(bnd, segid, nsegp);

  // 3) segment encoder (D=64, H=2, FF=128, seg mask)
  for (int l = 0; l < 2; ++l)
    run_layer(xs, xh_s, 64, 2, 128, 0,
              seg_awt[l][0], seg_awt[l][1], seg_awt[l][2], seg_awt[l][3],
              seg_attn_b + (size_t)(l * 4 + 0) * 64, seg_attn_b + (size_t)(l * 4 + 1) * 64,
              seg_attn_b + (size_t)(l * 4 + 2) * 64, seg_attn_b + (size_t)(l * 4 + 3) * 64,
              seg_ln_g + (size_t)(l * 2 + 0) * 64, seg_ln_b + (size_t)(l * 2 + 0) * 64,
              seg_ln_g + (size_t)(l * 2 + 1) * 64, seg_ln_b + (size_t)(l * 2 + 1) * 64,
              seg_f1t[l], seg_ff1_b + (size_t)l * 128, seg_f2t[l], seg_ff2_b + (size_t)l * 64);

  // 4) segment pooling + up-projection + validity mask
  fill0_kernel<<<(int)((M * 64 + 255) / 256), 256, 0, stream>>>(sums, M * 64);
  fill0_kernel<<<(int)((M + 255) / 256), 256, 0, stream>>>(cnts, M);
  scatter_kernel<<<(M_ROWS * 64 + 255) / 256, 256, 0, stream>>>(xs, segid, sums, cnts);
  mean_kernel<<<(M_ROWS * 64 + 255) / 256, 256, 0, stream>>>(sums, cnts, meanh);
  run_gemm(meanh, upwt, up_b, ybuf, nullptr, M_ROWS, 128, 64, 0);
  maskup_kernel<<<(M_ROWS * 128 + 255) / 256, 256, 0, stream>>>(ybuf, nsegp, xg, glbin, xh_g);

  // 5) global encoder (D=128, H=4, FF=256, valid mask)
  for (int l = 0; l < 2; ++l)
    run_layer(xg, xh_g, 128, 4, 256, 1,
              glb_awt[l][0], glb_awt[l][1], glb_awt[l][2], glb_awt[l][3],
              glb_attn_b + (size_t)(l * 4 + 0) * 128, glb_attn_b + (size_t)(l * 4 + 1) * 128,
              glb_attn_b + (size_t)(l * 4 + 2) * 128, glb_attn_b + (size_t)(l * 4 + 3) * 128,
              glb_ln_g + (size_t)(l * 2 + 0) * 128, glb_ln_b + (size_t)(l * 2 + 0) * 128,
              glb_ln_g + (size_t)(l * 2 + 1) * 128, glb_ln_b + (size_t)(l * 2 + 1) * 128,
              glb_f1t[l], glb_ff1_b + (size_t)l * 256, glb_f2t[l], glb_ff2_b + (size_t)l * 128);

  // 6) aggregate + final linear
  final_kernel<<<BATCH, 128, 0, stream>>>(xg, glbin, nsegp, fin_w, fin_b, (float*)d_out);
}
